// TemporalGNN_56899726737798
// MI455X (gfx1250) — matrix operand order repaired and checked by reading, not compiled, not run on hardware
//
#include <hip/hip_runtime.h>
#include <hip/hip_bf16.h>
#include <math.h>

typedef __attribute__((ext_vector_type(16))) __bf16 v16bf;
typedef __attribute__((ext_vector_type(8)))  float  v8f;

#define NEG_SLOPE 0.2f

// ---------------------------------------------------------------- utilities

__global__ void fill_kernel(float* __restrict__ p, long n, float v) {
    long i = (long)blockIdx.x * blockDim.x + threadIdx.x;
    if (i < n) p[i] = v;
}

// Pack W1[32,128] into WMMA-B register layout: out[(ntile*32+lane)*16 + i],
// element i of lane = B[k = (lane>>4)*16 + i][col = ntile*16 + (lane&15)]
__global__ void pack_w1_kernel(const float* __restrict__ W1, __bf16* __restrict__ out) {
    int idx = blockIdx.x * blockDim.x + threadIdx.x;
    if (idx >= 8 * 32 * 16) return;
    int i = idx & 15, lane = (idx >> 4) & 31, nt = idx >> 9;
    int k = ((lane >> 4) * 16) + i;
    int col = nt * 16 + (lane & 15);
    out[idx] = (__bf16)W1[k * 128 + col];
}

// Pack concat(K1,K2,K3) [256,96] into per-(kk,ntile) WMMA-B layout:
// out[((kk*6+nt)*32+lane)*16 + i]
__global__ void pack_kc_kernel(const float* __restrict__ K1, const float* __restrict__ K2,
                               const float* __restrict__ K3, __bf16* __restrict__ out) {
    int idx = blockIdx.x * blockDim.x + threadIdx.x;
    if (idx >= 8 * 6 * 32 * 16) return;
    int i = idx & 15, lane = (idx >> 4) & 31;
    int rest = idx >> 9;
    int nt = rest % 6, kk = rest / 6;
    int kg = kk * 32 + (lane >> 4) * 16 + i;     // 0..255
    int col = nt * 16 + (lane & 15);             // 0..95
    int conv = col >> 5, d = col & 31;
    int tkk = kg >> 7, c = kg & 127;
    const float* K = (conv == 0) ? K1 : ((conv == 1) ? K2 : K3);
    out[idx] = (__bf16)K[tkk * 4096 + c * 32 + d];
}

// Pack W2[32,16] into WMMA-B layout (single n-tile): out[lane*16 + i]
__global__ void pack_w2_kernel(const float* __restrict__ W2, __bf16* __restrict__ out) {
    int idx = blockIdx.x * blockDim.x + threadIdx.x;
    if (idx >= 32 * 16) return;
    int i = idx & 15, lane = idx >> 4;
    int k = (lane >> 4) * 16 + i;
    int col = lane & 15;
    out[idx] = (__bf16)W2[k * 16 + col];
}

__device__ __forceinline__ void atomicMaxFloat(float* addr, float val) {
    if (val >= 0.f) atomicMax((int*)addr, __float_as_int(val));
    else            atomicMin((unsigned int*)addr, __float_as_uint(val));
}

__device__ __forceinline__ void edge_decode(const int* __restrict__ src, const int* __restrict__ dst,
                                            int E, long e, int& s, int& d) {
    if (e < E) { s = src[e]; d = dst[e]; }
    else       { s = d = (int)(e - E); }      // implicit self-loop
}

// ---------------------------------------------------------------- WMMA GEMMs

__device__ __forceinline__ v8f wmma_bf16(v16bf a, v16bf b, v8f c) {
    return __builtin_amdgcn_wmma_f32_16x16x32_bf16(false, a, false, b, (short)0, c, false, false);
}

// Pack four float4s (two contiguous 8-float runs) into the 16-bf16 A operand.
__device__ __forceinline__ v16bf pack_a(float4 x0, float4 x1, float4 x2, float4 x3) {
    v16bf a;
    a[0]  = (__bf16)x0.x; a[1]  = (__bf16)x0.y; a[2]  = (__bf16)x0.z; a[3]  = (__bf16)x0.w;
    a[4]  = (__bf16)x1.x; a[5]  = (__bf16)x1.y; a[6]  = (__bf16)x1.z; a[7]  = (__bf16)x1.w;
    a[8]  = (__bf16)x2.x; a[9]  = (__bf16)x2.y; a[10] = (__bf16)x2.z; a[11] = (__bf16)x2.w;
    a[12] = (__bf16)x3.x; a[13] = (__bf16)x3.y; a[14] = (__bf16)x3.z; a[15] = (__bf16)x3.w;
    return a;
}

// Load A-tile operand for row `row` of a [*,32] fp32 matrix (K=32).
__device__ __forceinline__ v16bf load_a32(const float* __restrict__ ar, int klo) {
    const float4* a0 = (const float4*)(ar + 2 * klo);     // K = 2*klo .. 2*klo+7
    const float4* a1 = (const float4*)(ar + 2 * klo + 8); // K = 2*klo+8 .. 2*klo+15
    return pack_a(a0[0], a0[1], a1[0], a1[1]);
}

// h[N,128] = x[N,32] @ W1[32,128]  (one WMMA per 16x16 tile, K=32)
__global__ void gemm1_kernel(const float* __restrict__ x, const __bf16* __restrict__ wb,
                             float* __restrict__ h, int N) {
    const int lane = threadIdx.x & 31;
    const int wid  = threadIdx.x >> 5;     // ntile 0..7 (128 cols)
    const int m0   = blockIdx.x * 16;
    const int n0   = wid * 16;
    const int m    = lane & 15;
    const int klo  = (lane >> 4) * 8;
    int row = m0 + m; if (row >= N) row = N - 1;

    v16bf a = load_a32(x + (size_t)row * 32, klo);
    v16bf b = *(const v16bf*)(wb + ((size_t)wid * 32 + lane) * 16);
    v8f c = {};
    c = wmma_bf16(a, b, c);

    const int n  = lane & 15;
    const int mb = (lane >> 4) * 8;
    if (m0 + 16 <= N) {
#pragma unroll
        for (int v = 0; v < 8; ++v) h[(size_t)(m0 + mb + v) * 128 + n0 + n] = c[v];
    } else {
#pragma unroll
        for (int v = 0; v < 8; ++v) {
            int r = m0 + mb + v;
            if (r < N) h[(size_t)r * 128 + n0 + n] = c[v];
        }
    }
}

// conv_out[N,96] = window(xfeat)[N,256] @ Kc[256,96]
// window row n, col k -> xfeat[1+(k>>7)][n][k&127]; each 32-wide K block has uniform time slice
__global__ void temporal_gemm_kernel(const float* __restrict__ xf, const __bf16* __restrict__ kcb,
                                     float* __restrict__ co, int N) {
    const int lane = threadIdx.x & 31;
    const int wid  = threadIdx.x >> 5;     // ntile 0..5 (96 cols)
    const int m0   = blockIdx.x * 16;
    const int n0   = wid * 16;
    const int m    = lane & 15;
    const int klo  = (lane >> 4) * 8;
    int row = m0 + m; if (row >= N) row = N - 1;

    v8f c = {};
#pragma unroll
    for (int kk = 0; kk < 8; ++kk) {
        const float* ar = xf + ((size_t)(1 + (kk >> 2)) * N + row) * 128 + (kk & 3) * 32;
        v16bf a = load_a32(ar, klo);
        v16bf b = *(const v16bf*)(kcb + ((size_t)(kk * 6 + wid) * 32 + lane) * 16);
        c = wmma_bf16(a, b, c);
    }
    const int n  = lane & 15;
    const int mb = (lane >> 4) * 8;
    if (m0 + 16 <= N) {
#pragma unroll
        for (int v = 0; v < 8; ++v) co[(size_t)(m0 + mb + v) * 96 + n0 + n] = c[v];
    } else {
#pragma unroll
        for (int v = 0; v < 8; ++v) {
            int r = m0 + mb + v;
            if (r < N) co[(size_t)r * 96 + n0 + n] = c[v];
        }
    }
}

// h2[N,16] = xg[N,32] @ W2[32,16]; 8 m-tiles per block (one per wave)
__global__ void gemm2_kernel(const float* __restrict__ xg, const __bf16* __restrict__ wb,
                             float* __restrict__ h2, int N) {
    const int lane  = threadIdx.x & 31;
    const int wid   = threadIdx.x >> 5;
    const int mtile = blockIdx.x * 8 + wid;
    const int m0    = mtile * 16;
    if (m0 >= N) return;
    const int m   = lane & 15;
    const int klo = (lane >> 4) * 8;
    int row = m0 + m; if (row >= N) row = N - 1;

    v16bf a = load_a32(xg + (size_t)row * 32, klo);
    v16bf b = *(const v16bf*)(wb + (size_t)lane * 16);
    v8f c = {};
    c = wmma_bf16(a, b, c);

    const int n  = lane & 15;
    const int mb = (lane >> 4) * 8;
    if (m0 + 16 <= N) {
#pragma unroll
        for (int v = 0; v < 8; ++v) h2[(size_t)(m0 + mb + v) * 16 + n] = c[v];
    } else {
#pragma unroll
        for (int v = 0; v < 8; ++v) {
            int r = m0 + mb + v;
            if (r < N) h2[(size_t)r * 16 + n] = c[v];
        }
    }
}

// ---------------------------------------------------------------- GAT pieces

// as/ad[n*H+h] = dot(h[n,h,:], a_src/a_dst[h,:])
__global__ void scores_kernel(const float* __restrict__ h, const float* __restrict__ asrc,
                              const float* __restrict__ adst, float* __restrict__ as,
                              float* __restrict__ ad, int N, int H, int C) {
    int idx = blockIdx.x * blockDim.x + threadIdx.x;
    if (idx >= N * H) return;
    int n = idx / H, hh = idx % H;
    const float* hp = h + ((size_t)n * H + hh) * C;
    float s = 0.f, d = 0.f;
    for (int c = 0; c < C; ++c) { s += hp[c] * asrc[hh * C + c]; d += hp[c] * adst[hh * C + c]; }
    as[idx] = s; ad[idx] = d;
}

__global__ void edge_max_kernel(const int* __restrict__ src, const int* __restrict__ dst,
                                int E, int N, int H, const float* __restrict__ as,
                                const float* __restrict__ ad, float* __restrict__ m) {
    long idx = (long)blockIdx.x * blockDim.x + threadIdx.x;
    long tot = (long)(E + N) * H;
    if (idx >= tot) return;
    long e = idx / H; int hh = (int)(idx % H);
    int s, d; edge_decode(src, dst, E, e, s, d);
    float ev = as[(size_t)s * H + hh] + ad[(size_t)d * H + hh];
    ev = ev > 0.f ? ev : NEG_SLOPE * ev;
    atomicMaxFloat(&m[(size_t)d * H + hh], ev);
}

__global__ void edge_sum_kernel(const int* __restrict__ src, const int* __restrict__ dst,
                                int E, int N, int H, const float* __restrict__ as,
                                const float* __restrict__ ad, const float* __restrict__ m,
                                float* __restrict__ z) {
    long idx = (long)blockIdx.x * blockDim.x + threadIdx.x;
    long tot = (long)(E + N) * H;
    if (idx >= tot) return;
    long e = idx / H; int hh = (int)(idx % H);
    int s, d; edge_decode(src, dst, E, e, s, d);
    float ev = as[(size_t)s * H + hh] + ad[(size_t)d * H + hh];
    ev = ev > 0.f ? ev : NEG_SLOPE * ev;
    float p = __expf(ev - m[(size_t)d * H + hh]);
    atomicAdd(&z[(size_t)d * H + hh], p);
}

// Layer-1 aggregation: wave per edge, lane handles 4 channels (head = lane>>3)
__global__ void edge_agg1_kernel(const int* __restrict__ src, const int* __restrict__ dst,
                                 int E, int N, const float* __restrict__ as,
                                 const float* __restrict__ ad, const float* __restrict__ m,
                                 const float* __restrict__ z, const float* __restrict__ h1,
                                 float* __restrict__ out) {
    const int lane = threadIdx.x & 31;
    const int wid  = threadIdx.x >> 5;
    long e = (long)blockIdx.x * 8 + wid;
    if (e >= (long)E + N) return;
    int s, d; edge_decode(src, dst, E, e, s, d);
    int hh = lane >> 3;
    float ev = as[(size_t)s * 4 + hh] + ad[(size_t)d * 4 + hh];
    ev = ev > 0.f ? ev : NEG_SLOPE * ev;
    float alpha = __expf(ev - m[(size_t)d * 4 + hh]) / z[(size_t)d * 4 + hh];
    const float4 hv = *(const float4*)&h1[(size_t)s * 128 + lane * 4];
    float* o = &out[(size_t)d * 128 + lane * 4];
    atomicAdd(o + 0, alpha * hv.x);
    atomicAdd(o + 1, alpha * hv.y);
    atomicAdd(o + 2, alpha * hv.z);
    atomicAdd(o + 3, alpha * hv.w);
}

// Layer-2 aggregation: thread per (edge, channel<16)
__global__ void edge_agg2_kernel(const int* __restrict__ src, const int* __restrict__ dst,
                                 int E, int N, const float* __restrict__ as,
                                 const float* __restrict__ ad, const float* __restrict__ m,
                                 const float* __restrict__ z, const float* __restrict__ h2,
                                 float* __restrict__ agg) {
    long idx = (long)blockIdx.x * blockDim.x + threadIdx.x;
    long tot = (long)(E + N) * 16;
    if (idx >= tot) return;
    long e = idx >> 4; int c = (int)(idx & 15);
    int s, d; edge_decode(src, dst, E, e, s, d);
    float ev = as[s] + ad[d];
    ev = ev > 0.f ? ev : NEG_SLOPE * ev;
    float alpha = __expf(ev - m[d]) / z[d];
    atomicAdd(&agg[(size_t)d * 16 + c], alpha * h2[(size_t)s * 16 + c]);
}

__global__ void bias_relu_kernel(float* __restrict__ xt, const float* __restrict__ bias, long n) {
    long idx = (long)blockIdx.x * blockDim.x + threadIdx.x;
    if (idx >= n) return;
    int c = (int)(idx & 127);
    float v = xt[idx] + bias[c];
    xt[idx] = v > 0.f ? v : 0.f;
}

__global__ void gate_kernel(const float* __restrict__ co, const float* __restrict__ b1,
                            const float* __restrict__ b2, const float* __restrict__ b3,
                            float* __restrict__ xg, int N) {
    int idx = blockIdx.x * blockDim.x + threadIdx.x;
    if (idx >= N * 32) return;
    int n = idx >> 5, d = idx & 31;
    const float* r = co + (size_t)n * 96;
    float c1 = r[d] + b1[d];
    float c2 = r[32 + d] + b2[d];
    float c3 = r[64 + d] + b3[d];
    float g = c1 * (1.f / (1.f + __expf(-c2))) + c3;
    xg[idx] = g > 0.f ? g : 0.f;
}

// wave per node: 16-lane shuffle reduction for max / sum-exp
__global__ void logsoftmax_kernel(const float* __restrict__ agg, const float* __restrict__ bias,
                                  float* __restrict__ out, int N) {
    const int lane = threadIdx.x & 31;
    const int wid  = threadIdx.x >> 5;
    int node = blockIdx.x * 8 + wid;
    if (node >= N) return;
    int c = lane & 15;
    float v = agg[(size_t)node * 16 + c] + bias[c];
    float mx = v;
#pragma unroll
    for (int o = 1; o < 16; o <<= 1) mx = fmaxf(mx, __shfl_xor(mx, o));
    float se = __expf(v - mx);
#pragma unroll
    for (int o = 1; o < 16; o <<= 1) se += __shfl_xor(se, o);
    if (lane < 16) out[(size_t)node * 16 + c] = v - mx - __logf(se);
}

// ---------------------------------------------------------------- launcher

static inline int gdiv(long a, long b) { return (int)((a + b - 1) / b); }

extern "C" void kernel_launch(void* const* d_in, const int* in_sizes, int n_in,
                              void* d_out, int out_size, void* d_ws, size_t ws_size,
                              hipStream_t stream) {
    const int T = 3, HID = 32, HEADS = 4, OUT = 16;
    const int N = in_sizes[0] / (T * HID);
    const int E = in_sizes[1] / (T * 2);
    const long Etot = (long)E + N;

    const float* x_all = (const float*)d_in[0];
    const int*   ei    = (const int*)d_in[1];
    const float* W1    = (const float*)d_in[2];
    const float* as1w  = (const float*)d_in[3];
    const float* ad1w  = (const float*)d_in[4];
    const float* bias1 = (const float*)d_in[5];
    const float* K1    = (const float*)d_in[6];
    const float* bk1   = (const float*)d_in[7];
    const float* K2    = (const float*)d_in[8];
    const float* bk2   = (const float*)d_in[9];
    const float* K3    = (const float*)d_in[10];
    const float* bk3   = (const float*)d_in[11];
    const float* W2    = (const float*)d_in[12];
    const float* as2w  = (const float*)d_in[13];
    const float* ad2w  = (const float*)d_in[14];
    const float* bias2 = (const float*)d_in[15];
    float* out = (float*)d_out;

    // workspace layout (256B aligned)
    char* w = (char*)d_ws;
    size_t off = 0;
    auto alloc = [&](size_t bytes) { size_t o = off; off += (bytes + 255) & ~(size_t)255; return o; };
    float*  xfeat = (float*)(w + alloc((size_t)3 * N * 128 * 4));  // relu(GAT1) per t
    float*  h1    = (float*)(w + alloc((size_t)N * 128 * 4));      // reused: conv_out + xg
    float*  conv_out = h1;                                          // [N,96] overlays h1
    float*  xg    = h1 + (size_t)N * 96;                            // [N,32] overlays h1 tail
    float*  as1   = (float*)(w + alloc((size_t)N * 4 * 4));
    float*  ad1   = (float*)(w + alloc((size_t)N * 4 * 4));
    float*  m1    = (float*)(w + alloc((size_t)N * 4 * 4));
    float*  z1    = (float*)(w + alloc((size_t)N * 4 * 4));
    float*  h2    = (float*)(w + alloc((size_t)N * 16 * 4));
    float*  as2   = (float*)(w + alloc((size_t)N * 4));
    float*  ad2   = (float*)(w + alloc((size_t)N * 4));
    float*  m2    = (float*)(w + alloc((size_t)N * 4));
    float*  z2    = (float*)(w + alloc((size_t)N * 4));
    float*  agg2  = (float*)(w + alloc((size_t)N * 16 * 4));
    __bf16* W1bf  = (__bf16*)(w + alloc((size_t)8 * 32 * 16 * 2));      // packed B layout
    __bf16* Kcbf  = (__bf16*)(w + alloc((size_t)8 * 6 * 32 * 16 * 2));  // packed B layout
    __bf16* W2bf  = (__bf16*)(w + alloc((size_t)32 * 16 * 2));          // packed B layout
    (void)ws_size; (void)n_in; (void)out_size;

    const float NINF = -__builtin_inff();

    // weight conversions into WMMA-B register layouts
    pack_w1_kernel<<<gdiv(8 * 32 * 16, 256), 256, 0, stream>>>(W1, W1bf);
    pack_kc_kernel<<<gdiv(8 * 6 * 32 * 16, 256), 256, 0, stream>>>(K1, K2, K3, Kcbf);
    pack_w2_kernel<<<gdiv(32 * 16, 256), 256, 0, stream>>>(W2, W2bf);

    const int mtiles = gdiv(N, 16);

    // ---- per-snapshot GAT layer 1 ----
    for (int t = 0; t < T; ++t) {
        const float* xt    = x_all + (size_t)t * N * 32;
        const int*   src_t = ei + (size_t)t * 2 * E;
        const int*   dst_t = src_t + E;
        float*       xft   = xfeat + (size_t)t * N * 128;

        fill_kernel<<<gdiv((long)N * 128, 256), 256, 0, stream>>>(xft, (long)N * 128, 0.f);
        fill_kernel<<<gdiv((long)N * 4, 256), 256, 0, stream>>>(m1, (long)N * 4, NINF);
        fill_kernel<<<gdiv((long)N * 4, 256), 256, 0, stream>>>(z1, (long)N * 4, 0.f);

        gemm1_kernel<<<mtiles, 256, 0, stream>>>(xt, W1bf, h1, N);
        scores_kernel<<<gdiv((long)N * HEADS, 256), 256, 0, stream>>>(h1, as1w, ad1w, as1, ad1, N, HEADS, HID);
        edge_max_kernel<<<gdiv(Etot * HEADS, 256), 256, 0, stream>>>(src_t, dst_t, E, N, HEADS, as1, ad1, m1);
        edge_sum_kernel<<<gdiv(Etot * HEADS, 256), 256, 0, stream>>>(src_t, dst_t, E, N, HEADS, as1, ad1, m1, z1);
        edge_agg1_kernel<<<gdiv(Etot, 8), 256, 0, stream>>>(src_t, dst_t, E, N, as1, ad1, m1, z1, h1, xft);
        bias_relu_kernel<<<gdiv((long)N * 128, 256), 256, 0, stream>>>(xft, bias1, (long)N * 128);
    }

    // ---- temporal conv (only t_out = T-2 feeds the head) ----
    temporal_gemm_kernel<<<mtiles, 192, 0, stream>>>(xfeat, Kcbf, conv_out, N);
    gate_kernel<<<gdiv((long)N * 32, 256), 256, 0, stream>>>(conv_out, bk1, bk2, bk3, xg, N);

    // ---- GAT layer 2 on last snapshot's edges ----
    const int* src_l = ei + (size_t)(T - 1) * 2 * E;
    const int* dst_l = src_l + E;

    fill_kernel<<<gdiv((long)N, 256), 256, 0, stream>>>(m2, N, NINF);
    fill_kernel<<<gdiv((long)N, 256), 256, 0, stream>>>(z2, N, 0.f);
    fill_kernel<<<gdiv((long)N * 16, 256), 256, 0, stream>>>(agg2, (long)N * 16, 0.f);

    gemm2_kernel<<<gdiv(mtiles, 8), 256, 0, stream>>>(xg, W2bf, h2, N);
    scores_kernel<<<gdiv((long)N, 256), 256, 0, stream>>>(h2, as2w, ad2w, as2, ad2, N, 1, OUT);
    edge_max_kernel<<<gdiv(Etot, 256), 256, 0, stream>>>(src_l, dst_l, E, N, 1, as2, ad2, m2);
    edge_sum_kernel<<<gdiv(Etot, 256), 256, 0, stream>>>(src_l, dst_l, E, N, 1, as2, ad2, m2, z2);
    edge_agg2_kernel<<<gdiv(Etot * 16, 256), 256, 0, stream>>>(src_l, dst_l, E, N, as2, ad2, m2, z2, h2, agg2);

    logsoftmax_kernel<<<gdiv(N, 8), 256, 0, stream>>>(agg2, bias2, out, N);
}
